// ARCLLMUnified_89498528514807
// MI455X (gfx1250) — matrix operand order repaired and checked by reading, not compiled, not run on hardware
//
#include <hip/hip_runtime.h>
#include <math.h>

// Problem constants (from reference)
#define DV 32000
#define DD 1024
#define DL 2
#define DH 16
#define DR 16
#define DHD 64
#define DB 2
#define DT 1024
#define DM (DB * DT)
#define DF (4 * DD)

typedef __attribute__((ext_vector_type(16))) __bf16 v16bf;
typedef __attribute__((ext_vector_type(8))) float v8f;

__device__ __forceinline__ unsigned short f2bf(float x) {
  unsigned int u = __float_as_uint(x);
  u += 0x7FFFu + ((u >> 16) & 1u);  // round-to-nearest-even
  return (unsigned short)(u >> 16);
}

union FragU { uint4 u[2]; v16bf v; };

// A fragment (16x32 bf16): lanes 0-15 hold K {0..7,16..23}, lanes 16-31 hold K {8..15,24..31}
__device__ __forceinline__ v16bf load_frag_a(const unsigned short* p) {
  FragU f;
  f.u[0] = *(const uint4*)p;
  f.u[1] = *(const uint4*)(p + 16);
  return f.v;
}
// B fragment from B^T (N x K row-major): lanes 0-15 hold K {0..7,16..23}, lanes 16-31 hold K {8..15,24..31}
__device__ __forceinline__ v16bf load_frag_b(const unsigned short* p) {
  FragU f;
  f.u[0] = *(const uint4*)p;
  f.u[1] = *(const uint4*)(p + 16);
  return f.v;
}

// Generic batched bf16 WMMA GEMM: C = act(A @ B + bias) [+ resid]
// A: M x K bf16 row-major (lda). Bt: N x K bf16 row-major (ldb) == B transposed.
// Register blocking: each wave computes 64x32 of C (4 M-tiles x 2 N-tiles);
// the 4 waves of a block stack in M (256 rows) sharing the same 32 B columns.
// Output: f32 (Cf) or bf16 (Cb), row stride ldc, batch z offset = (z/nH)*cStrideB + (z%nH)*cStrideH.
// flags bit0: exact GELU.
__global__ __launch_bounds__(128) void wmma_gemm_kernel(
    const unsigned short* __restrict__ A, const unsigned short* __restrict__ Bt,
    const float* __restrict__ bias, const float* __restrict__ resid,
    float* __restrict__ Cf, unsigned short* __restrict__ Cb,
    int M, int N, int K, int lda, int ldb, int ldc,
    long aBatch, long bBatch, long cStrideB, long cStrideH, int nH, int flags) {
  const int lane = threadIdx.x & 31;
  const int wave = threadIdx.x >> 5;
  const int nBlk = N >> 5;   // 32-column blocks
  const int mBlk = M >> 8;   // 256-row macro blocks (4 waves x 64 rows)
  const int blk = blockIdx.x;
  if (blk >= mBlk * nBlk) return;
  const int m0 = ((blk / nBlk) << 8) + (wave << 6);
  const int n0 = (blk % nBlk) << 5;
  const int z = blockIdx.y;
  const unsigned short* Az = A + (long)z * aBatch;
  const unsigned short* Bz = Bt + (long)z * bBatch;
  const long cOff = (long)(z / nH) * cStrideB + (long)(z % nH) * cStrideH;

  const int r16 = lane & 15;
  const int hi = lane >> 4;
  const long aRow0 = (long)(m0 + r16) * lda + (hi << 3);
  const long bRow0 = (long)(n0 + r16) * ldb + (hi << 3);
  const long aStep = (long)16 * lda;
  const long bStep = (long)16 * ldb;

  v8f acc[4][2];
#pragma unroll
  for (int mi = 0; mi < 4; ++mi)
#pragma unroll
    for (int ni = 0; ni < 2; ++ni) acc[mi][ni] = (v8f){0.f,0.f,0.f,0.f,0.f,0.f,0.f,0.f};

  for (int k0 = 0; k0 < K; k0 += 32) {
    // pull k-data 1024 elements ahead in both streams toward the caches (global_prefetch_b8)
    __builtin_prefetch(Az + aRow0 + k0 + 1024, 0, 1);
    __builtin_prefetch(Bz + bRow0 + k0 + 1024, 0, 1);
    const v16bf b0 = load_frag_b(Bz + bRow0 + k0);
    const v16bf b1 = load_frag_b(Bz + bRow0 + bStep + k0);
#pragma unroll
    for (int mi = 0; mi < 4; ++mi) {
      const v16bf a = load_frag_a(Az + aRow0 + (long)mi * aStep + k0);
      acc[mi][0] = __builtin_amdgcn_wmma_f32_16x16x32_bf16(false, a, false, b0, (short)0,
                                                           acc[mi][0], false, false);
      acc[mi][1] = __builtin_amdgcn_wmma_f32_16x16x32_bf16(false, a, false, b1, (short)0,
                                                           acc[mi][1], false, false);
    }
  }

#pragma unroll
  for (int ni = 0; ni < 2; ++ni) {
    const int col = n0 + (ni << 4) + r16;
    const float bv = bias ? bias[col] : 0.f;
#pragma unroll
    for (int mi = 0; mi < 4; ++mi) {
      const int rbase = m0 + (mi << 4) + (hi << 3);
#pragma unroll
      for (int i = 0; i < 8; ++i) {
        float x = acc[mi][ni][i] + bv;
        if (flags & 1) x = 0.5f * x * (1.f + erff(x * 0.70710678118654752f));
        const long idx = cOff + (long)(rbase + i) * ldc + col;
        if (resid) x += resid[idx];
        if (Cb) Cb[idx] = f2bf(x);
        else    Cf[idx] = x;
      }
    }
  }
}

__global__ __launch_bounds__(256) void embed_kernel(const int* __restrict__ ids,
                                                    const float* __restrict__ emb,
                                                    float* __restrict__ x) {
  const int row = blockIdx.x;
  const long src = (long)ids[row] * DD;
  const long dst = (long)row * DD;
  for (int i = threadIdx.x; i < DD; i += 256) x[dst + i] = emb[src + i];
}

// LayerNorm over D=1024, writes bf16
__global__ __launch_bounds__(256) void ln_kernel(const float* __restrict__ x,
                                                 const float* __restrict__ w,
                                                 const float* __restrict__ b,
                                                 unsigned short* __restrict__ out) {
  __shared__ float red[256];
  const int tid = threadIdx.x;
  const long base = (long)blockIdx.x * DD;
  float v[4];
  float s = 0.f;
#pragma unroll
  for (int i = 0; i < 4; ++i) { v[i] = x[base + tid + 256 * i]; s += v[i]; }
  red[tid] = s; __syncthreads();
  for (int st = 128; st > 0; st >>= 1) { if (tid < st) red[tid] += red[tid + st]; __syncthreads(); }
  const float mu = red[0] * (1.f / DD);
  __syncthreads();
  float var = 0.f;
#pragma unroll
  for (int i = 0; i < 4; ++i) { const float d = v[i] - mu; var += d * d; }
  red[tid] = var; __syncthreads();
  for (int st = 128; st > 0; st >>= 1) { if (tid < st) red[tid] += red[tid + st]; __syncthreads(); }
  const float rs = rsqrtf(red[0] * (1.f / DD) + 1e-5f);
#pragma unroll
  for (int i = 0; i < 4; ++i) {
    const int j = tid + 256 * i;
    out[base + j] = f2bf((v[i] - mu) * rs * w[j] + b[j]);
  }
}

// Per-head metric g[h] = A[h]^T A[h] + exp(log_lambda[h]) * I  (64x64)
__global__ __launch_bounds__(256) void gmetric_kernel(const float* __restrict__ Ah,
                                                      const float* __restrict__ loglam,
                                                      float* __restrict__ g) {
  const int h = blockIdx.x;
  const float lam = __expf(loglam[h]);
  const float* Ab = Ah + (long)h * DR * DHD;
  for (int idx = threadIdx.x; idx < DHD * DHD; idx += 256) {
    const int d = idx >> 6, e = idx & 63;
    float s = 0.f;
#pragma unroll
    for (int r = 0; r < DR; ++r) s += Ab[r * DHD + d] * Ab[r * DHD + e];
    if (d == e) s += lam;
    g[(long)h * DHD * DHD + idx] = s;
  }
}

// For each (b,h,t): xg = x @ g[h] (bf16, (B,H,T,64) layout); optionally raw x bf16 copy
// and kk = x^T g x (used for K path).
__global__ __launch_bounds__(64) void metric_kernel(const float* __restrict__ xin,
                                                    const float* __restrict__ g,
                                                    unsigned short* __restrict__ outg,
                                                    unsigned short* __restrict__ outraw,
                                                    float* __restrict__ kkout) {
  __shared__ float sx[64];
  __shared__ float sr[64];
  const int d = threadIdx.x;
  long idx = blockIdx.x;
  const int t = (int)(idx % DT); idx /= DT;
  const int h = (int)(idx % DH); const int b = (int)(idx / DH);
  const float* xr = xin + ((long)b * DT + t) * DD + h * DHD;
  sx[d] = xr[d];
  __syncthreads();
  const float* gh = g + (long)h * DHD * DHD;
  float acc = 0.f;
#pragma unroll 8
  for (int e = 0; e < DHD; ++e) acc += sx[e] * gh[e * DHD + d];
  const long o = (((long)b * DH + h) * DT + t) * DHD + d;
  if (outg)   outg[o]   = f2bf(acc);
  if (outraw) outraw[o] = f2bf(sx[d]);
  if (kkout) {
    sr[d] = acc * sx[d];
    __syncthreads();
    for (int st = 32; st > 0; st >>= 1) { if (d < st) sr[d] += sr[d + st]; __syncthreads(); }
    if (d == 0) kkout[((long)b * DH + h) * DT + t] = sr[0];
  }
}

// V (B,T,D) f32 -> Vt (B,H,64,T) bf16 for the P@V GEMM's B^T operand
__global__ __launch_bounds__(256) void vtrans_kernel(const float* __restrict__ v,
                                                     unsigned short* __restrict__ vt) {
  long i = (long)blockIdx.x * 256 + threadIdx.x;
  const int s = (int)(i & (DT - 1)); long r = i >> 10;
  const int d = (int)(r & (DHD - 1)); r >>= 6;
  const int h = (int)(r & (DH - 1)); const int b = (int)(r >> 4);
  vt[i] = f2bf(v[((long)b * DT + s) * DD + h * DHD + d]);
}

// softmax over s of (2*S[t,s] - kk[s]); writes bf16 P
__global__ __launch_bounds__(256) void softmax_kernel(const float* __restrict__ S,
                                                      const float* __restrict__ kk,
                                                      unsigned short* __restrict__ P) {
  __shared__ float red[256];
  const long row = blockIdx.x;          // bh*T + t
  const long bh = row / DT;
  const float* sr = S + row * DT;
  const float* kr = kk + bh * DT;
  const int tid = threadIdx.x;
  float v[4];
  float mx = -1e30f;
#pragma unroll
  for (int i = 0; i < 4; ++i) {
    const int j = tid + 256 * i;
    const float l = 2.f * sr[j] - kr[j];
    v[i] = l;
    mx = fmaxf(mx, l);
  }
  red[tid] = mx; __syncthreads();
  for (int st = 128; st > 0; st >>= 1) { if (tid < st) red[tid] = fmaxf(red[tid], red[tid + st]); __syncthreads(); }
  mx = red[0];
  __syncthreads();
  float sum = 0.f;
#pragma unroll
  for (int i = 0; i < 4; ++i) { v[i] = __expf(v[i] - mx); sum += v[i]; }
  red[tid] = sum; __syncthreads();
  for (int st = 128; st > 0; st >>= 1) { if (tid < st) red[tid] += red[tid + st]; __syncthreads(); }
  const float inv = 1.f / red[0];
  unsigned short* pr = P + row * DT;
#pragma unroll
  for (int i = 0; i < 4; ++i) pr[tid + 256 * i] = f2bf(v[i] * inv);
}

// f32 (K x N) row-major -> bf16 (N x K) row-major (transpose-convert)
__global__ __launch_bounds__(256) void convt_kernel(const float* __restrict__ in,
                                                    unsigned short* __restrict__ out,
                                                    int K, int N) {
  const long i = (long)blockIdx.x * 256 + threadIdx.x;
  if (i >= (long)K * N) return;
  const int n = (int)(i % N);
  const int k = (int)(i / N);
  out[(long)n * K + k] = f2bf(in[i]);
}

extern "C" void kernel_launch(void* const* d_in, const int* in_sizes, int n_in,
                              void* d_out, int out_size, void* d_ws, size_t ws_size,
                              hipStream_t stream) {
  (void)in_sizes; (void)n_in; (void)out_size; (void)ws_size;
  const int*   ids    = (const int*)d_in[0];
  const float* embed  = (const float*)d_in[1];
  const float* Aw     = (const float*)d_in[2];
  const float* loglam = (const float*)d_in[3];
  const float* wq = (const float*)d_in[4];  const float* bq = (const float*)d_in[5];
  const float* wk = (const float*)d_in[6];  const float* bk = (const float*)d_in[7];
  const float* wv = (const float*)d_in[8];  const float* bv = (const float*)d_in[9];
  const float* wo = (const float*)d_in[10]; const float* bo = (const float*)d_in[11];
  const float* ln1w = (const float*)d_in[12]; const float* ln1b = (const float*)d_in[13];
  const float* ln2w = (const float*)d_in[14]; const float* ln2b = (const float*)d_in[15];
  const float* w1 = (const float*)d_in[16]; const float* b1 = (const float*)d_in[17];
  const float* w2 = (const float*)d_in[18]; const float* b2 = (const float*)d_in[19];
  const float* normfw = (const float*)d_in[20]; const float* normfb = (const float*)d_in[21];
  const float* headw  = (const float*)d_in[22]; const float* headb  = (const float*)d_in[23];
  float* out = (float*)d_out;

  // Workspace carve-out
  char* pws = (char*)d_ws;
  auto wsAlloc = [&](size_t bytes) -> void* {
    void* r = (void*)pws;
    pws += (bytes + 255) & ~(size_t)255;
    return r;
  };
  unsigned short* wqt = (unsigned short*)wsAlloc((size_t)DL * DD * DD * 2);
  unsigned short* wkt = (unsigned short*)wsAlloc((size_t)DL * DD * DD * 2);
  unsigned short* wvt = (unsigned short*)wsAlloc((size_t)DL * DD * DD * 2);
  unsigned short* wot = (unsigned short*)wsAlloc((size_t)DL * DD * DD * 2);
  unsigned short* w1t = (unsigned short*)wsAlloc((size_t)DL * DD * DF * 2);
  unsigned short* w2t = (unsigned short*)wsAlloc((size_t)DL * DF * DD * 2);
  unsigned short* headt = (unsigned short*)wsAlloc((size_t)DD * DV * 2);
  float* xA = (float*)wsAlloc((size_t)DM * DD * 4);
  float* xB = (float*)wsAlloc((size_t)DM * DD * 4);
  unsigned short* hb = (unsigned short*)wsAlloc((size_t)DM * DD * 2);
  float* qf = (float*)wsAlloc((size_t)DM * DD * 4);
  float* kf = (float*)wsAlloc((size_t)DM * DD * 4);
  float* vf = (float*)wsAlloc((size_t)DM * DD * 4);
  float* gbuf = (float*)wsAlloc((size_t)DH * DHD * DHD * 4);
  unsigned short* qgb = (unsigned short*)wsAlloc((size_t)DB * DH * DT * DHD * 2);
  unsigned short* kb  = (unsigned short*)wsAlloc((size_t)DB * DH * DT * DHD * 2);
  float* kkf = (float*)wsAlloc((size_t)DB * DH * DT * 4);
  unsigned short* vtb = (unsigned short*)wsAlloc((size_t)DB * DH * DHD * DT * 2);
  float* Sbuf = (float*)wsAlloc((size_t)DB * DH * DT * DT * 4);
  unsigned short* Pb = (unsigned short*)wsAlloc((size_t)DB * DH * DT * DT * 2);
  unsigned short* interb = (unsigned short*)wsAlloc((size_t)DM * DF * 2);
  unsigned short* ob = (unsigned short*)wsAlloc((size_t)DM * DD * 2);
  unsigned short* fb = (unsigned short*)wsAlloc((size_t)DM * DD * 2);

  auto gemm = [&](const unsigned short* A, const unsigned short* Bt, const float* bias,
                  const float* resid, float* Cf, unsigned short* Cb, int M, int N, int K,
                  int lda, int ldb, int ldc, long aB, long bB, long cB, long cH, int nH,
                  int nbatch, int flags) {
    const int blocks = (M / 256) * (N / 32);  // one 256x32 macro-tile per block (4 waves)
    dim3 grid(blocks, nbatch);
    wmma_gemm_kernel<<<grid, dim3(128), 0, stream>>>(A, Bt, bias, resid, Cf, Cb, M, N, K,
                                                     lda, ldb, ldc, aB, bB, cB, cH, nH, flags);
  };
  auto convT = [&](const float* in, unsigned short* outp, int K, int N) {
    const long tot = (long)K * N;
    convt_kernel<<<dim3((unsigned)((tot + 255) / 256)), dim3(256), 0, stream>>>(in, outp, K, N);
  };

  // Convert all weights to bf16, transposed (N x K) for the B^T operand
  for (int l = 0; l < DL; ++l) {
    convT(wq + (long)l * DD * DD, wqt + (long)l * DD * DD, DD, DD);
    convT(wk + (long)l * DD * DD, wkt + (long)l * DD * DD, DD, DD);
    convT(wv + (long)l * DD * DD, wvt + (long)l * DD * DD, DD, DD);
    convT(wo + (long)l * DD * DD, wot + (long)l * DD * DD, DD, DD);
    convT(w1 + (long)l * DD * DF, w1t + (long)l * DD * DF, DD, DF);
    convT(w2 + (long)l * DF * DD, w2t + (long)l * DF * DD, DF, DD);
  }
  convT(headw, headt, DD, DV);

  // Embedding gather
  embed_kernel<<<dim3(DM), dim3(256), 0, stream>>>(ids, embed, xA);

  for (int l = 0; l < DL; ++l) {
    const long wOff = (long)l * DD * DD;
    // ln1 -> bf16
    ln_kernel<<<dim3(DM), dim3(256), 0, stream>>>(xA, ln1w + l * DD, ln1b + l * DD, hb);
    // q,k,v projections (f32 out)
    gemm(hb, wqt + wOff, bq + l * DD, nullptr, qf, nullptr, DM, DD, DD, DD, DD, DD,
         0, 0, 0, 0, 1, 1, 0);
    gemm(hb, wkt + wOff, bk + l * DD, nullptr, kf, nullptr, DM, DD, DD, DD, DD, DD,
         0, 0, 0, 0, 1, 1, 0);
    gemm(hb, wvt + wOff, bv + l * DD, nullptr, vf, nullptr, DM, DD, DD, DD, DD, DD,
         0, 0, 0, 0, 1, 1, 0);
    // per-head metric g, then qg / (k bf16 + kk) / V^T
    gmetric_kernel<<<dim3(DH), dim3(256), 0, stream>>>(Aw + (long)l * DH * DR * DHD,
                                                       loglam + l * DH, gbuf);
    metric_kernel<<<dim3(DB * DH * DT), dim3(64), 0, stream>>>(qf, gbuf, qgb, nullptr, nullptr);
    metric_kernel<<<dim3(DB * DH * DT), dim3(64), 0, stream>>>(kf, gbuf, nullptr, kb, kkf);
    vtrans_kernel<<<dim3((DB * DH * DHD * DT) / 256), dim3(256), 0, stream>>>(vf, vtb);
    // S = QG @ K^T  (batched over b*h)
    gemm(qgb, kb, nullptr, nullptr, Sbuf, nullptr, DT, DT, DHD, DHD, DHD, DT,
         (long)DT * DHD, (long)DT * DHD, (long)DT * DT, 0, 1, DB * DH, 0);
    // P = softmax(2S - kk) -> bf16
    softmax_kernel<<<dim3(DB * DH * DT), dim3(256), 0, stream>>>(Sbuf, kkf, Pb);
    // O = P @ V -> bf16 (b,t,D) concat-head layout
    gemm(Pb, vtb, nullptr, nullptr, nullptr, ob, DT, DHD, DT, DT, DT, DD,
         (long)DT * DT, (long)DHD * DT, (long)DT * DD, DHD, DH, DB * DH, 0);
    // x = x + O @ wo + bo
    gemm(ob, wot + wOff, bo + l * DD, xA, xB, nullptr, DM, DD, DD, DD, DD, DD,
         0, 0, 0, 0, 1, 1, 0);
    // FFN
    ln_kernel<<<dim3(DM), dim3(256), 0, stream>>>(xB, ln2w + l * DD, ln2b + l * DD, hb);
    gemm(hb, w1t + (long)l * DD * DF, b1 + l * DF, nullptr, nullptr, interb, DM, DF, DD,
         DD, DD, DF, 0, 0, 0, 0, 1, 1, /*gelu*/ 1);
    gemm(interb, w2t + (long)l * DF * DD, b2 + l * DD, xB, xA, nullptr, DM, DD, DF,
         DF, DF, DD, 0, 0, 0, 0, 1, 1, 0);
  }

  // Final LN + LM head
  ln_kernel<<<dim3(DM), dim3(256), 0, stream>>>(xA, normfw, normfb, fb);
  gemm(fb, headt, headb, nullptr, out, nullptr, DM, DV, DD, DD, DD, DV,
       0, 0, 0, 0, 1, 1, 0);
}
